// WindowedAttentionBlockNd_78778290143736
// MI455X (gfx1250) — matrix operand order repaired and checked by reading, not compiled, not run on hardware
//
#include <hip/hip_runtime.h>

// ---------------------------------------------------------------------------
// Swin-style shifted-window attention block for MI455X (gfx1250, wave32).
// One workgroup (8 waves) per window; all GEMMs via v_wmma_f32_16x16x32_bf16.
// Phase-1 residual copy uses GLOBAL_LOAD_ASYNC_TO_LDS_B128 (ASYNCcnt-tracked).
// ---------------------------------------------------------------------------

typedef unsigned short u16;
typedef __attribute__((ext_vector_type(16))) __bf16 bf16x16;
typedef __attribute__((ext_vector_type(8)))  float  f32x8;
typedef int gv4i __attribute__((vector_size(16)));   // matches builtin's V4i

union FragU { bf16x16 v; uint4 q[2]; };

#define BATCH 32
#define HDIM 64
#define WDIM 64
#define CDIM 256
#define KWIN 8
#define SHIFT 4
#define TOK 64          // tokens per window
#define NWIN 64         // windows per image
#define HEADS 8
#define DHEAD 32
#define HID 768

#if defined(__HIP_DEVICE_COMPILE__) && __has_builtin(__builtin_amdgcn_global_load_async_to_lds_b128)
#define HAVE_ASYNC_LDS 1
#else
#define HAVE_ASYNC_LDS 0
#endif

__device__ __forceinline__ u16 f2bf(float f) {
    unsigned int u = __float_as_uint(f);
    unsigned int r = (u + 0x7FFFu + ((u >> 16) & 1u)) >> 16;   // RNE
    return (u16)r;
}

__device__ __forceinline__ f32x8 fzero() {
    f32x8 z;
    #pragma unroll
    for (int i = 0; i < 8; ++i) z[i] = 0.0f;
    return z;
}

__device__ __forceinline__ f32x8 wmma_bf16(bf16x16 a, bf16x16 b, f32x8 c) {
    // (neg_a, A, neg_b, B, c_mod, C, reuse_a, reuse_b)
    return __builtin_amdgcn_wmma_f32_16x16x32_bf16(false, a, false, b, (short)0, c, false, false);
}

#if HAVE_ASYNC_LDS
// 16B global -> LDS via the async datamover (ASYNCcnt-tracked, no VGPR data).
__device__ __forceinline__ void async_copy16(const float* g, float* l) {
    __builtin_amdgcn_global_load_async_to_lds_b128(
        (__attribute__((address_space(1))) gv4i*)(void*)const_cast<float*>(g),
        (__attribute__((address_space(3))) gv4i*)(void*)l,
        0 /*offset*/, 0 /*cpol*/);
}
#endif

__device__ __forceinline__ void wait_async0() {
#if HAVE_ASYNC_LDS
#if __has_builtin(__builtin_amdgcn_s_wait_asynccnt)
    __builtin_amdgcn_s_wait_asynccnt(0);
#else
    asm volatile("s_wait_asynccnt 0" ::: "memory");
#endif
#endif
}

// A fragment (16x32 bf16): lanes 0-15 row M=lane, K chunks {k0..k0+7, k0+16..k0+23};
// lanes 16-31 same rows, K chunks {k0+8.., k0+24..}. Two 16B LDS/global loads.
__device__ __forceinline__ bf16x16 load_a(const u16* base, int row, int ld, int k0, int lane) {
    int kb = k0 + ((lane & 16) >> 1);
    const u16* p = base + row * ld + kb;
    FragU f;
    f.q[0] = *(const uint4*)(p);
    f.q[1] = *(const uint4*)(p + 16);
    return f.v;
}

// B fragment (32x16 bf16), source stored [N][K] row-major: lane n holds column
// n (lanes 0-15: K chunks {k0..k0+7, k0+16..k0+23}, lanes 16-31: {k0+8.., k0+24..}).
__device__ __forceinline__ bf16x16 load_b(const u16* base, int col, int ld, int k0, int lane) {
    int kb = k0 + ((lane & 16) >> 1);
    const u16* p = base + (size_t)col * ld + kb;
    FragU f;
    f.q[0] = *(const uint4*)(p);
    f.q[1] = *(const uint4*)(p + 16);
    return f.v;
}

__device__ __forceinline__ float gelu_tanh(float x) {
    const float c0 = 0.7978845608028654f;   // sqrt(2/pi)
    const float c1 = 0.044715f;
    return 0.5f * x * (1.0f + tanhf(c0 * (x + c1 * x * x * x)));
}

// ---------------------------------------------------------------------------
// Pre-pass: convert weights fp32 -> bf16, transposed to [out][in] row-major.
// ---------------------------------------------------------------------------
__global__ void prep_weights_kernel(const float* __restrict__ wq, const float* __restrict__ wk,
                                    const float* __restrict__ wv, const float* __restrict__ wo,
                                    const float* __restrict__ w1, const float* __restrict__ w2,
                                    u16* __restrict__ wqT, u16* __restrict__ wkT,
                                    u16* __restrict__ wvT, u16* __restrict__ woT,
                                    u16* __restrict__ w1T, u16* __restrict__ w2T) {
    int idx = blockIdx.x * 256 + threadIdx.x;
    if (idx < 65536) {                       // wqT[256][256] <- wq[256][256]
        int o = idx >> 8, i = idx & 255;
        wqT[idx] = f2bf(wq[i * 256 + o]);
    } else if (idx < 131072) {
        int j = idx - 65536; int o = j >> 8, i = j & 255;
        wkT[j] = f2bf(wk[i * 256 + o]);
    } else if (idx < 196608) {
        int j = idx - 131072; int o = j >> 8, i = j & 255;
        wvT[j] = f2bf(wv[i * 256 + o]);
    } else if (idx < 262144) {
        int j = idx - 196608; int o = j >> 8, i = j & 255;
        woT[j] = f2bf(wo[i * 256 + o]);
    } else if (idx < 458752) {               // w1T[768][256] <- w1[256][768]
        int j = idx - 262144; int o = j >> 8, i = j & 255;
        w1T[j] = f2bf(w1[i * 768 + o]);
    } else if (idx < 655360) {               // w2T[256][768] <- w2[768][256]
        int j = idx - 458752; int o = j / 768, i = j % 768;
        w2T[j] = f2bf(w2[i * 256 + o]);
    }
}

// bias8[h][t][s] = rpe_table[rel_index[t][s]][h]
__global__ void prep_bias_kernel(const float* __restrict__ rpe, const int* __restrict__ rel,
                                 float* __restrict__ bias8) {
    int idx = blockIdx.x * 256 + threadIdx.x;   // 8*64*64 = 32768
    if (idx < HEADS * TOK * TOK) {
        int h = idx >> 12;
        int ts = idx & 4095;
        bias8[idx] = rpe[rel[ts] * HEADS + h];
    }
}

// ---------------------------------------------------------------------------
// Main fused kernel: one workgroup per window.
// LDS: xf f32[64][256] (64K) | qb/kb/vT bf16 (96K) | union U (96K) = 256K.
// ---------------------------------------------------------------------------
__global__ __launch_bounds__(256) void swin_block_kernel(
    const float* __restrict__ query,
    const float* __restrict__ bq, const float* __restrict__ bk, const float* __restrict__ bv,
    const float* __restrict__ bo,
    const float* __restrict__ ln1g, const float* __restrict__ ln1b,
    const float* __restrict__ ln2g, const float* __restrict__ ln2b,
    const float* __restrict__ b1, const float* __restrict__ b2,
    const float* __restrict__ mask,         // [NWIN][64][64]
    const u16* __restrict__ wqT, const u16* __restrict__ wkT,
    const u16* __restrict__ wvT, const u16* __restrict__ woT,
    const u16* __restrict__ w1T, const u16* __restrict__ w2T,
    const float* __restrict__ bias8,        // [HEADS][64][64]
    float* __restrict__ out)
{
    extern __shared__ char smem[];
    float* xf = (float*)smem;                               // [64][256] fp32 residual
    u16*   qb = (u16*)(smem + 65536);                       // [64][256]
    u16*   kb = qb + TOK * CDIM;                            // [64][256]
    u16*   vT = kb + TOK * CDIM;                            // [256][64] (transposed V)
    char*  U  = smem + 65536 + 3 * TOK * CDIM * 2;
    u16*   xb    = (u16*)U;                                 // [64][256] bf16 input
    u16*   attnb = (u16*)U;                                 // [8][64][64] softmax probs
    float* oo    = (float*)U;                               // [64][256] o@wo fp32
    u16*   hid   = (u16*)U;                                 // [64][768] MLP hidden
    u16*   ob    = qb;                                      // attn output (reuses q)
    u16*   x1b   = qb;                                      // bf16 of LN1 out (reuses q)
    float* out2  = (float*)qb;                              // MLP2 fp32 (reuses q/k)

    const int tid  = threadIdx.x;
    const int lane = tid & 31;
    const int wid  = tid >> 5;                              // 8 waves
    const int blk  = blockIdx.x;
    const int bimg = blk >> 6;
    const int nw   = blk & 63;
    const int wy   = nw >> 3, wx = nw & 7;
    const int l15  = lane & 15;
    const int hlf  = (lane & 16) >> 1;                      // 0 / 8

    // ---- Phase 1: gather shifted window -------------------------------
    // fp32 residual -> xf via async global->LDS datamover (ASYNCcnt);
    // bf16 working copy -> xb via the regular VGPR path (overlapped).
    for (int i = tid; i < TOK * 64; i += 256) {             // 64 float4 per row
        int row = i >> 6, c4 = (i & 63) << 2;
        int ky = row >> 3, kx = row & 7;
        int hh = (wy * KWIN + ky + SHIFT) & 63;
        int ww = (wx * KWIN + kx + SHIFT) & 63;
        const float* gp = &query[(((size_t)bimg * HDIM + hh) * WDIM + ww) * CDIM + c4];
        float* lp = &xf[row * CDIM + c4];
#if HAVE_ASYNC_LDS
        async_copy16(gp, lp);
#endif
        float4 v = *(const float4*)gp;
#if !HAVE_ASYNC_LDS
        *(float4*)lp = v;
#endif
        u16* xp = &xb[row * CDIM + c4];
        xp[0] = f2bf(v.x); xp[1] = f2bf(v.y); xp[2] = f2bf(v.z); xp[3] = f2bf(v.w);
    }
    wait_async0();
    __syncthreads();

    // ---- Phase 2: QKV projections (x[64,256] @ W[256,256] + b) ----------
    for (int mat = 0; mat < 3; ++mat) {
        const u16* wT = (mat == 0) ? wqT : (mat == 1) ? wkT : wvT;
        const float* bias = (mat == 0) ? bq : (mat == 1) ? bk : bv;
        for (int nn = 0; nn < 2; ++nn) {
            int nt = wid * 2 + nn;
            int col = nt * 16 + l15;
            f32x8 acc[4];
            #pragma unroll
            for (int mt = 0; mt < 4; ++mt) acc[mt] = fzero();
            for (int kk = 0; kk < 8; ++kk) {
                bf16x16 bfrag = load_b(wT, col, 256, kk * 32, lane);
                #pragma unroll
                for (int mt = 0; mt < 4; ++mt) {
                    bf16x16 afrag = load_a(xb, mt * 16 + l15, 256, kk * 32, lane);
                    acc[mt] = wmma_bf16(afrag, bfrag, acc[mt]);
                }
            }
            float bia = bias[col];
            #pragma unroll
            for (int mt = 0; mt < 4; ++mt) {
                int rbase = mt * 16 + hlf;
                #pragma unroll
                for (int r = 0; r < 8; ++r) {
                    float v = acc[mt][r] + bia;
                    int row = rbase + r;
                    if (mat == 0)      qb[row * CDIM + col] = f2bf(v);
                    else if (mat == 1) kb[row * CDIM + col] = f2bf(v);
                    else               vT[col * TOK + row]  = f2bf(v);   // transposed
                }
            }
        }
    }
    __syncthreads();

    // ---- Phase 3: logits (q@k^T), softmax in registers, head = wave -----
    {
        const int h = wid;
        const float scale = 0.17677669529663687f;   // 1/sqrt(32)
        for (int mt = 0; mt < 4; ++mt) {
            f32x8 acc[4];
            #pragma unroll
            for (int nt = 0; nt < 4; ++nt) {
                bf16x16 a = load_a(qb, mt * 16 + l15, 256, h * 32, lane);
                bf16x16 b = load_b(kb, nt * 16 + l15, 256, h * 32, lane);
                acc[nt] = wmma_bf16(a, b, fzero());
            }
            #pragma unroll
            for (int r = 0; r < 8; ++r) {
                int m = mt * 16 + hlf + r;
                float l[4];
                #pragma unroll
                for (int nt = 0; nt < 4; ++nt) {
                    int n = nt * 16 + l15;
                    l[nt] = acc[nt][r] * scale
                          + bias8[(h * TOK + m) * TOK + n]
                          + mask[((size_t)nw * TOK + m) * TOK + n];
                }
                float mx = fmaxf(fmaxf(l[0], l[1]), fmaxf(l[2], l[3]));
                for (int s = 1; s < 16; s <<= 1) mx = fmaxf(mx, __shfl_xor(mx, s));
                float p[4], sum = 0.0f;
                #pragma unroll
                for (int nt = 0; nt < 4; ++nt) { p[nt] = __expf(l[nt] - mx); sum += p[nt]; }
                for (int s = 1; s < 16; s <<= 1) sum += __shfl_xor(sum, s);
                float inv = 1.0f / sum;
                #pragma unroll
                for (int nt = 0; nt < 4; ++nt)
                    attnb[(h * TOK + m) * TOK + nt * 16 + l15] = f2bf(p[nt] * inv);
            }
        }
    }
    __syncthreads();

    // ---- Phase 4: attn @ V  ->  o[64, DHEAD] per head -------------------
    {
        const int h = wid;
        const u16* ab = attnb + h * TOK * TOK;
        for (int nt = 0; nt < 2; ++nt) {
            int col = h * DHEAD + nt * 16 + l15;
            for (int mt = 0; mt < 4; ++mt) {
                f32x8 acc = fzero();
                #pragma unroll
                for (int ks = 0; ks < 2; ++ks) {
                    bf16x16 a = load_a(ab, mt * 16 + l15, TOK, ks * 32, lane);
                    bf16x16 b = load_b(vT, col, TOK, ks * 32, lane);
                    acc = wmma_bf16(a, b, acc);
                }
                int rbase = mt * 16 + hlf;
                #pragma unroll
                for (int r = 0; r < 8; ++r)
                    ob[(rbase + r) * CDIM + col] = f2bf(acc[r]);
            }
        }
    }
    __syncthreads();

    // ---- Phase 5: o @ wo + bo -> oo (fp32) ------------------------------
    for (int nn = 0; nn < 2; ++nn) {
        int nt = wid * 2 + nn;
        int col = nt * 16 + l15;
        f32x8 acc[4];
        #pragma unroll
        for (int mt = 0; mt < 4; ++mt) acc[mt] = fzero();
        for (int kk = 0; kk < 8; ++kk) {
            bf16x16 bfrag = load_b(woT, col, 256, kk * 32, lane);
            #pragma unroll
            for (int mt = 0; mt < 4; ++mt) {
                bf16x16 afrag = load_a(ob, mt * 16 + l15, 256, kk * 32, lane);
                acc[mt] = wmma_bf16(afrag, bfrag, acc[mt]);
            }
        }
        float bia = bo[col];
        #pragma unroll
        for (int mt = 0; mt < 4; ++mt) {
            int rbase = mt * 16 + hlf;
            #pragma unroll
            for (int r = 0; r < 8; ++r)
                oo[(rbase + r) * CDIM + col] = acc[mt][r] + bia;
        }
    }
    __syncthreads();

    // ---- Phase 6: LN1 + residual; x1 -> xf (fp32) and x1b (bf16) --------
    for (int rr = 0; rr < 8; ++rr) {
        int row = wid * 8 + rr;
        float vals[8], s = 0.0f, s2 = 0.0f;
        #pragma unroll
        for (int j = 0; j < 8; ++j) {
            float v = oo[row * CDIM + lane + 32 * j];
            vals[j] = v; s += v; s2 += v * v;
        }
        for (int m = 1; m < 32; m <<= 1) { s += __shfl_xor(s, m); s2 += __shfl_xor(s2, m); }
        float mean = s * (1.0f / 256.0f);
        float var  = s2 * (1.0f / 256.0f) - mean * mean;
        float rs   = rsqrtf(var + 1e-5f);
        #pragma unroll
        for (int j = 0; j < 8; ++j) {
            int c = lane + 32 * j;
            float y  = (vals[j] - mean) * rs * ln1g[c] + ln1b[c];
            float x1 = y + xf[row * CDIM + c];
            xf[row * CDIM + c]  = x1;
            x1b[row * CDIM + c] = f2bf(x1);
        }
    }
    __syncthreads();

    // ---- Phase 7: MLP up: gelu(x1 @ w1 + b1) -> hid bf16 [64][768] ------
    for (int nn = 0; nn < 6; ++nn) {
        int nt = wid * 6 + nn;
        int col = nt * 16 + l15;
        f32x8 acc[4];
        #pragma unroll
        for (int mt = 0; mt < 4; ++mt) acc[mt] = fzero();
        for (int kk = 0; kk < 8; ++kk) {
            bf16x16 bfrag = load_b(w1T, col, 256, kk * 32, lane);
            #pragma unroll
            for (int mt = 0; mt < 4; ++mt) {
                bf16x16 afrag = load_a(x1b, mt * 16 + l15, 256, kk * 32, lane);
                acc[mt] = wmma_bf16(afrag, bfrag, acc[mt]);
            }
        }
        float bia = b1[col];
        #pragma unroll
        for (int mt = 0; mt < 4; ++mt) {
            int rbase = mt * 16 + hlf;
            #pragma unroll
            for (int r = 0; r < 8; ++r)
                hid[(rbase + r) * HID + col] = f2bf(gelu_tanh(acc[mt][r] + bia));
        }
    }
    __syncthreads();

    // ---- Phase 8: MLP down: hid @ w2 + b2 -> out2 fp32 ------------------
    for (int nn = 0; nn < 2; ++nn) {
        int nt = wid * 2 + nn;
        int col = nt * 16 + l15;
        f32x8 acc[4];
        #pragma unroll
        for (int mt = 0; mt < 4; ++mt) acc[mt] = fzero();
        for (int kk = 0; kk < 24; ++kk) {
            bf16x16 bfrag = load_b(w2T, col, 768, kk * 32, lane);
            #pragma unroll
            for (int mt = 0; mt < 4; ++mt) {
                bf16x16 afrag = load_a(hid, mt * 16 + l15, 768, kk * 32, lane);
                acc[mt] = wmma_bf16(afrag, bfrag, acc[mt]);
            }
        }
        float bia = b2[col];
        #pragma unroll
        for (int mt = 0; mt < 4; ++mt) {
            int rbase = mt * 16 + hlf;
            #pragma unroll
            for (int r = 0; r < 8; ++r)
                out2[(rbase + r) * CDIM + col] = acc[mt][r] + bia;
        }
    }
    __syncthreads();

    // ---- Phase 9: LN2 + residual, scatter with unshift ------------------
    for (int rr = 0; rr < 8; ++rr) {
        int row = wid * 8 + rr;
        float vals[8], s = 0.0f, s2 = 0.0f;
        #pragma unroll
        for (int j = 0; j < 8; ++j) {
            float v = out2[row * CDIM + lane + 32 * j];
            vals[j] = v; s += v; s2 += v * v;
        }
        for (int m = 1; m < 32; m <<= 1) { s += __shfl_xor(s, m); s2 += __shfl_xor(s2, m); }
        float mean = s * (1.0f / 256.0f);
        float var  = s2 * (1.0f / 256.0f) - mean * mean;
        float rs   = rsqrtf(var + 1e-5f);
        int ky = row >> 3, kx = row & 7;
        int hh = (wy * KWIN + ky + SHIFT) & 63;
        int ww = (wx * KWIN + kx + SHIFT) & 63;
        size_t gbase = (((size_t)bimg * HDIM + hh) * WDIM + ww) * CDIM;
        #pragma unroll
        for (int j = 0; j < 8; ++j) {
            int c = lane + 32 * j;
            float y = (vals[j] - mean) * rs * ln2g[c] + ln2b[c];
            out[gbase + c] = y + xf[row * CDIM + c];
        }
    }
}

// ---------------------------------------------------------------------------
extern "C" void kernel_launch(void* const* d_in, const int* in_sizes, int n_in,
                              void* d_out, int out_size, void* d_ws, size_t ws_size,
                              hipStream_t stream) {
    const float* query = (const float*)d_in[0];
    const float* wq = (const float*)d_in[1];  const float* bq = (const float*)d_in[2];
    const float* wk = (const float*)d_in[3];  const float* bk = (const float*)d_in[4];
    const float* wv = (const float*)d_in[5];  const float* bv = (const float*)d_in[6];
    const float* wo = (const float*)d_in[7];  const float* bo = (const float*)d_in[8];
    const float* rpe = (const float*)d_in[9];
    const float* ln1g = (const float*)d_in[10]; const float* ln1b = (const float*)d_in[11];
    const float* ln2g = (const float*)d_in[12]; const float* ln2b = (const float*)d_in[13];
    const float* w1 = (const float*)d_in[14]; const float* b1 = (const float*)d_in[15];
    const float* w2 = (const float*)d_in[16]; const float* b2 = (const float*)d_in[17];
    const float* attn_mask = (const float*)d_in[18];
    const int*   rel_index = (const int*)d_in[19];

    // Workspace layout (bf16 transposed weights, then fp32 rpe bias).
    u16* ws  = (u16*)d_ws;
    u16* wqT = ws;
    u16* wkT = ws + 65536;
    u16* wvT = ws + 131072;
    u16* woT = ws + 196608;
    u16* w1T = ws + 262144;       // [768][256]
    u16* w2T = ws + 458752;       // [256][768]
    float* bias8 = (float*)(ws + 655360);   // byte offset 1310720, 16B aligned

    prep_weights_kernel<<<(655360 + 255) / 256, 256, 0, stream>>>(
        wq, wk, wv, wo, w1, w2, wqT, wkT, wvT, woT, w1T, w2T);
    prep_bias_kernel<<<(HEADS * TOK * TOK + 255) / 256, 256, 0, stream>>>(
        rpe, rel_index, bias8);

    const int smem_bytes = 65536 + 3 * TOK * CDIM * 2 + 98304;   // 262144
    (void)hipFuncSetAttribute((const void*)swin_block_kernel,
                              hipFuncAttributeMaxDynamicSharedMemorySize, smem_bytes);

    swin_block_kernel<<<BATCH * NWIN, 256, smem_bytes, stream>>>(
        query, bq, bk, bv, bo, ln1g, ln1b, ln2g, ln2b, b1, b2,
        attn_mask, wqT, wkT, wvT, woT, w1T, w2T, bias8, (float*)d_out);
}
